// Encoder_layer_42064909697654
// MI455X (gfx1250) — matrix operand order repaired and checked by reading, not compiled, not run on hardware
//
#include <hip/hip_runtime.h>
#include <hip/hip_bf16.h>
#include <math.h>

typedef __bf16 bf16;
typedef __attribute__((ext_vector_type(16))) __bf16 v16bf;
typedef __attribute__((ext_vector_type(8)))  float  v8f;

#define D_MODEL  1024
#define NUM_HEAD 16
#define D_HEAD   64
#define D_FFN    4096
#define SEQ      2048
#define BATCH    2
#define NTOK     (BATCH * SEQ)   // 4096 tokens
#define LN_EPS   1e-6f

union Frag16 { v16bf v; uint4 u[2]; };

static __device__ __forceinline__ v8f zero8() {
  v8f z = {0.f, 0.f, 0.f, 0.f, 0.f, 0.f, 0.f, 0.f};
  return z;
}

static __device__ __forceinline__ v8f wmma_bf16(v16bf a, v16bf b, v8f c) {
  // 8 args: (neg_a, A, neg_b, B, c_mod, C, reuse_a, reuse_b)
  return __builtin_amdgcn_wmma_f32_16x16x32_bf16(false, a, false, b, (short)0, c,
                                                 false, false);
}

// CDNA5 async copy: 32 bytes global -> LDS per call (2 x b128, ASYNCcnt += 2).
// INST_OFFSET is added to BOTH the LDS and global address (ISA 08 section 4.4).
static __device__ __forceinline__ void async_copy_32B(unsigned lds_addr,
                                                      const bf16* gptr) {
  asm volatile(
      "global_load_async_to_lds_b128 %0, %1, off\n\t"
      "global_load_async_to_lds_b128 %0, %1, off offset:16"
      :: "v"(lds_addr), "v"(gptr) : "memory");
}

static __device__ __forceinline__ void wait_async_le4() {
  asm volatile("s_wait_asynccnt 0x4" ::: "memory");
}
static __device__ __forceinline__ void wait_async_0() {
  asm volatile("s_wait_asynccnt 0x0" ::: "memory");
}
static __device__ __forceinline__ void wait_ds_0() {
  asm volatile("s_wait_dscnt 0x0" ::: "memory");
}

static __device__ __forceinline__ unsigned lds_off(const void* p) {
  // generic pointer to __shared__: low 32 bits are the LDS byte address
  return (unsigned)(size_t)p;
}

// ---------------------------------------------------------------------------
// fp32 -> bf16 conversion (vectorized by 4)
// ---------------------------------------------------------------------------
__global__ void cvt_bf16_kernel(const float* __restrict__ in,
                                bf16* __restrict__ out, int n4) {
  int i = blockIdx.x * blockDim.x + threadIdx.x;
  if (i < n4) {
    float4 f = ((const float4*)in)[i];
    bf16* o = out + 4 * (size_t)i;
    o[0] = (bf16)f.x; o[1] = (bf16)f.y; o[2] = (bf16)f.z; o[3] = (bf16)f.w;
  }
}

// ---------------------------------------------------------------------------
// C[M,N] = act( A[M,K] @ B[N,K]^T + bias[N] )   (A,B bf16, C fp32 and/or bf16)
// Block tile 128x128, 8 waves, each wave a 32x64 subtile.
// Double-buffered LDS fed by async global->LDS copies (ASYNCcnt pipeline).
// ---------------------------------------------------------------------------
__global__ __launch_bounds__(256) void gemm_wmma_kernel(
    const bf16* __restrict__ A, const bf16* __restrict__ B,
    const float* __restrict__ bias, float* __restrict__ outF,
    bf16* __restrict__ outB, int M, int N, int K, int relu) {
  __shared__ bf16 lds_a[2][128][40];   // 32 cols + 8 pad (bank stagger)
  __shared__ bf16 lds_b[2][128][40];

  const int tid  = threadIdx.x;
  const int lane = tid & 31;
  const int wave = tid >> 5;
  const int wm   = wave & 3;        // 4 wave-rows  x 32
  const int wn   = wave >> 2;       // 2 wave-cols  x 64
  const int row0 = blockIdx.y * 128;
  const int col0 = blockIdx.x * 128;

  const int ldr = tid >> 1;         // staging row 0..127
  const int ldc = (tid & 1) * 16;   // staging col half: 16 bf16 = 32B

  const int lm    = lane & 15;
  const int hi    = (lane & 16) ? 1 : 0;
  const int klo_a = hi ? 16 : 0;    // A-frag: lo lanes K 0-15, hi 16-31
  const int klo_b = hi ? 16 : 0;    // B-frag: lo lanes K 0-15, hi 16-31

  const bf16* gA = A + (size_t)(row0 + ldr) * K + ldc;
  const bf16* gB = B + (size_t)(col0 + ldr) * K + ldc;

  v8f acc[2][4];
#pragma unroll
  for (int m = 0; m < 2; ++m)
#pragma unroll
    for (int n = 0; n < 4; ++n) acc[m][n] = zero8();

  // issue tile 0 (4 async b128 per thread)
  async_copy_32B(lds_off(&lds_a[0][ldr][ldc]), gA);
  async_copy_32B(lds_off(&lds_b[0][ldr][ldc]), gB);

  const int nk = K >> 5;
  for (int kt = 0; kt < nk; ++kt) {
    const int cur = kt & 1;
    if (kt + 1 < nk) {
      const int nxt = cur ^ 1;
      async_copy_32B(lds_off(&lds_a[nxt][ldr][ldc]), gA + (kt + 1) * 32);
      async_copy_32B(lds_off(&lds_b[nxt][ldr][ldc]), gB + (kt + 1) * 32);
      wait_async_le4();   // tile kt landed (in-order), tile kt+1 in flight
    } else {
      wait_async_0();
    }
    __syncthreads();

    // ---- load fragments, contiguous k (lane half h: k 16h..16h+15) ----
    Frag16 af[2], bf_[4];
#pragma unroll
    for (int m = 0; m < 2; ++m) {
      const int r = wm * 32 + m * 16 + lm;
      af[m].u[0] = *(const uint4*)&lds_a[cur][r][klo_a];
      af[m].u[1] = *(const uint4*)&lds_a[cur][r][klo_a + 8];
    }
#pragma unroll
    for (int n = 0; n < 4; ++n) {
      const int c = wn * 64 + n * 16 + lm;
      bf_[n].u[0] = *(const uint4*)&lds_b[cur][c][klo_b];
      bf_[n].u[1] = *(const uint4*)&lds_b[cur][c][klo_b + 8];
    }
#pragma unroll
    for (int m = 0; m < 2; ++m)
#pragma unroll
      for (int n = 0; n < 4; ++n)
        acc[m][n] = wmma_bf16(af[m].v, bf_[n].v, acc[m][n]);

    if (kt + 1 < nk) __syncthreads();  // release buf cur for tile kt+2 writes
  }

  // ---- epilogue: C layout lane=col(lm), vgpr r = row r (lo) / r+8 (hi) ----
#pragma unroll
  for (int n = 0; n < 4; ++n) {
    const int c = col0 + wn * 64 + n * 16 + lm;
    const float bv = bias ? bias[c] : 0.f;
#pragma unroll
    for (int m = 0; m < 2; ++m) {
#pragma unroll
      for (int r = 0; r < 8; ++r) {
        const int rr = row0 + wm * 32 + m * 16 + r + hi * 8;
        float v = acc[m][n][r] + bv;
        if (relu) v = fmaxf(v, 0.f);
        if (outF) outF[(size_t)rr * N + c] = v;
        if (outB) outB[(size_t)rr * N + c] = (bf16)v;
      }
    }
  }
}

// ---------------------------------------------------------------------------
// Flash attention: q = k = v = hb (bf16 [NTOK][D_MODEL]).
// One wave per (batch, head, 16-query tile); online softmax over 2048 keys.
// V tile is staged via async global->LDS, issued ahead of the score WMMAs.
// ---------------------------------------------------------------------------
__global__ __launch_bounds__(32) void flash_attn_kernel(
    const bf16* __restrict__ hb, float* __restrict__ att) {
  __shared__ bf16 p_s[16][32];      // probabilities re-layout (C -> A frag)
  __shared__ bf16 v_s[32][64];      // staged V tile (32 keys x 64 dims)

  const int lane = threadIdx.x;
  const int lm   = lane & 15;
  const int hi   = (lane & 16) ? 1 : 0;
  const int qt   = blockIdx.x;              // 0..127
  const int bh   = blockIdx.y;              // 0..31
  const int bidx = bh >> 4;
  const int head = bh & 15;
  const int tok0 = bidx * SEQ;
  const int q0   = tok0 + qt * 16;
  const int hoff = head * D_HEAD;

  const int klo_a = hi ? 16 : 0;
  const int klo_b = hi ? 16 : 0;

  // Q fragments: two K-steps over d (0-31, 32-63)
  Frag16 qa[2];
  {
    const bf16* qrow = hb + (size_t)(q0 + lm) * D_MODEL + hoff;
    qa[0].u[0] = *(const uint4*)(qrow + klo_a);
    qa[0].u[1] = *(const uint4*)(qrow + klo_a + 8);
    qa[1].u[0] = *(const uint4*)(qrow + 32 + klo_a);
    qa[1].u[1] = *(const uint4*)(qrow + 32 + klo_a + 8);
  }

  v8f acc[4];
#pragma unroll
  for (int nd = 0; nd < 4; ++nd) acc[nd] = zero8();
  float mrow[8], lrow[8];
#pragma unroll
  for (int r = 0; r < 8; ++r) { mrow[r] = -1e30f; lrow[r] = 0.f; }

  const unsigned v_lds = lds_off(&v_s[lane][0]);

  for (int kt = 0; kt < SEQ; kt += 32) {
    const int k0 = tok0 + kt;

    // ---- issue V tile async copy early (128B per lane = 8 x b128) ----
    wait_ds_0();   // prior iteration's v_s ds_loads complete before overwrite
    {
      const bf16* vsrc = hb + (size_t)(k0 + lane) * D_MODEL + hoff;
      asm volatile(
          "global_load_async_to_lds_b128 %0, %1, off\n\t"
          "global_load_async_to_lds_b128 %0, %1, off offset:16\n\t"
          "global_load_async_to_lds_b128 %0, %1, off offset:32\n\t"
          "global_load_async_to_lds_b128 %0, %1, off offset:48\n\t"
          "global_load_async_to_lds_b128 %0, %1, off offset:64\n\t"
          "global_load_async_to_lds_b128 %0, %1, off offset:80\n\t"
          "global_load_async_to_lds_b128 %0, %1, off offset:96\n\t"
          "global_load_async_to_lds_b128 %0, %1, off offset:112"
          :: "v"(v_lds), "v"(vsrc) : "memory");
    }

    // ---- scores: S = (Q K^T) * 1/8, 16q x 32k, K-dim = d = 64 ----
    v8f s[2];
#pragma unroll
    for (int nt = 0; nt < 2; ++nt) {
      const bf16* krow = hb + (size_t)(k0 + nt * 16 + lm) * D_MODEL + hoff;
      Frag16 kf0, kf1;
      kf0.u[0] = *(const uint4*)(krow + klo_b);
      kf0.u[1] = *(const uint4*)(krow + klo_b + 8);
      kf1.u[0] = *(const uint4*)(krow + 32 + klo_b);
      kf1.u[1] = *(const uint4*)(krow + 32 + klo_b + 8);
      v8f z = zero8();
      z = wmma_bf16(qa[0].v, kf0.v, z);
      z = wmma_bf16(qa[1].v, kf1.v, z);
      s[nt] = z;
    }

    // ---- online softmax, rowwise across 16 lanes of each half-wave ----
#pragma unroll
    for (int r = 0; r < 8; ++r) {
      float a0 = s[0][r] * 0.125f;
      float a1 = s[1][r] * 0.125f;
      float mx = fmaxf(a0, a1);
      mx = fmaxf(mx, __shfl_xor(mx, 1, 32));
      mx = fmaxf(mx, __shfl_xor(mx, 2, 32));
      mx = fmaxf(mx, __shfl_xor(mx, 4, 32));
      mx = fmaxf(mx, __shfl_xor(mx, 8, 32));
      const float mn = fmaxf(mrow[r], mx);
      const float al = __expf(mrow[r] - mn);
      const float p0 = __expf(a0 - mn);
      const float p1 = __expf(a1 - mn);
      float ps = p0 + p1;
      ps += __shfl_xor(ps, 1, 32);
      ps += __shfl_xor(ps, 2, 32);
      ps += __shfl_xor(ps, 4, 32);
      ps += __shfl_xor(ps, 8, 32);
      lrow[r] = lrow[r] * al + ps;
      mrow[r] = mn;
#pragma unroll
      for (int nd = 0; nd < 4; ++nd) acc[nd][r] = acc[nd][r] * al;
      const int pr = r + hi * 8;
      p_s[pr][lm]      = (bf16)p0;
      p_s[pr][lm + 16] = (bf16)p1;
    }

    // ---- V tile must be resident before fragment gathers ----
    wait_async_0();
    __syncthreads();

    // ---- P-fragment (A layout) from LDS ----
    Frag16 pf;
    pf.u[0] = *(const uint4*)&p_s[lm][klo_a];
    pf.u[1] = *(const uint4*)&p_s[lm][klo_a + 8];

    // ---- O += P V : 4 N-tiles over d ----
#pragma unroll
    for (int nd = 0; nd < 4; ++nd) {
      Frag16 vf;
      const int dc = nd * 16 + lm;
#pragma unroll
      for (int j = 0; j < 8; ++j) {
        vf.v[2 * j]     = v_s[klo_b + 2 * j][dc];
        vf.v[2 * j + 1] = v_s[klo_b + 2 * j + 1][dc];
      }
      acc[nd] = wmma_bf16(pf.v, vf.v, acc[nd]);
    }
    __syncthreads();
  }

  // ---- normalize and write fp32 attention output ----
#pragma unroll
  for (int nd = 0; nd < 4; ++nd) {
#pragma unroll
    for (int r = 0; r < 8; ++r) {
      const int rr = q0 + r + hi * 8;
      att[(size_t)rr * D_MODEL + hoff + nd * 16 + lm] = acc[nd][r] / lrow[r];
    }
  }
}

// ---------------------------------------------------------------------------
// out = LayerNorm(x + y) with ddof=1 std and (std + eps) denominator.
// One block per token, 256 threads, 4 elements each.
// ---------------------------------------------------------------------------
__global__ __launch_bounds__(256) void add_ln_kernel(
    const float* __restrict__ x, const float* __restrict__ y,
    const float* __restrict__ g, const float* __restrict__ b,
    float* __restrict__ outF, bf16* __restrict__ outB) {
  __shared__ float red_s[8];
  __shared__ float red_q[8];
  __shared__ float stats[2];

  const int row = blockIdx.x;
  const int tid = threadIdx.x;
  const size_t base = (size_t)row * D_MODEL;

  float v[4];
  float s = 0.f, q = 0.f;
#pragma unroll
  for (int i = 0; i < 4; ++i) {
    const int c = i * 256 + tid;
    v[i] = x[base + c] + y[base + c];
    s += v[i];
    q += v[i] * v[i];
  }
  for (int mask = 1; mask < 32; mask <<= 1) {
    s += __shfl_xor(s, mask, 32);
    q += __shfl_xor(q, mask, 32);
  }
  if ((tid & 31) == 0) { red_s[tid >> 5] = s; red_q[tid >> 5] = q; }
  __syncthreads();
  if (tid == 0) {
    float ts = 0.f, tq = 0.f;
#pragma unroll
    for (int w = 0; w < 8; ++w) { ts += red_s[w]; tq += red_q[w]; }
    const float mean = ts * (1.f / D_MODEL);
    const float var  = fmaxf(tq - ts * mean, 0.f) * (1.f / (D_MODEL - 1));
    stats[0] = mean;
    stats[1] = sqrtf(var) + LN_EPS;
  }
  __syncthreads();
  const float mean = stats[0];
  const float rden = 1.f / stats[1];
#pragma unroll
  for (int i = 0; i < 4; ++i) {
    const int c = i * 256 + tid;
    const float o = g[c] * ((v[i] - mean) * rden) + b[c];
    outF[base + c] = o;
    if (outB) outB[base + c] = (bf16)o;
  }
}

// ---------------------------------------------------------------------------
extern "C" void kernel_launch(void* const* d_in, const int* in_sizes, int n_in,
                              void* d_out, int out_size, void* d_ws, size_t ws_size,
                              hipStream_t stream) {
  (void)in_sizes; (void)n_in; (void)out_size; (void)ws_size;

  const float* x    = (const float*)d_in[0];
  const float* wq   = (const float*)d_in[1];
  const float* bq   = (const float*)d_in[2];
  const float* ln1a = (const float*)d_in[3];
  const float* ln1b = (const float*)d_in[4];
  const float* ln2a = (const float*)d_in[5];
  const float* ln2b = (const float*)d_in[6];
  const float* w1   = (const float*)d_in[7];
  const float* b1   = (const float*)d_in[8];
  const float* w2   = (const float*)d_in[9];
  const float* b2   = (const float*)d_in[10];
  float* out = (float*)d_out;

  char* p = (char*)d_ws;
  auto alloc = [&](size_t bytes) -> char* {
    char* r = p;
    p += (bytes + 255) & ~(size_t)255;
    return r;
  };
  bf16*  xb  = (bf16*) alloc((size_t)NTOK * D_MODEL * 2);
  bf16*  wqb = (bf16*) alloc((size_t)D_MODEL * D_MODEL * 2);
  bf16*  w1b = (bf16*) alloc((size_t)D_FFN * D_MODEL * 2);
  bf16*  w2b = (bf16*) alloc((size_t)D_MODEL * D_FFN * 2);
  bf16*  hb  = (bf16*) alloc((size_t)NTOK * D_MODEL * 2);
  float* att = (float*)alloc((size_t)NTOK * D_MODEL * 4);
  float* x1  = (float*)alloc((size_t)NTOK * D_MODEL * 4);
  bf16*  x1b = (bf16*) alloc((size_t)NTOK * D_MODEL * 2);
  bf16*  y1b = (bf16*) alloc((size_t)NTOK * D_FFN * 2);
  float* f2  = (float*)alloc((size_t)NTOK * D_MODEL * 4);

  auto cvt = [&](const float* src, bf16* dst, size_t n) {
    const int n4 = (int)(n / 4);
    cvt_bf16_kernel<<<(n4 + 255) / 256, 256, 0, stream>>>(src, dst, n4);
  };
  cvt(x,  xb,  (size_t)NTOK * D_MODEL);
  cvt(wq, wqb, (size_t)D_MODEL * D_MODEL);
  cvt(w1, w1b, (size_t)D_FFN * D_MODEL);
  cvt(w2, w2b, (size_t)D_MODEL * D_FFN);

  // h = x @ wq^T + bq  (bf16 out for attention)
  gemm_wmma_kernel<<<dim3(D_MODEL / 128, NTOK / 128), 256, 0, stream>>>(
      xb, wqb, bq, nullptr, hb, NTOK, D_MODEL, D_MODEL, 0);

  // attention (q = k = v = h)
  flash_attn_kernel<<<dim3(SEQ / 16, BATCH * NUM_HEAD), 32, 0, stream>>>(hb, att);

  // x1 = LN1(x + att); bf16 copy for FFN
  add_ln_kernel<<<NTOK, 256, 0, stream>>>(x, att, ln1a, ln1b, x1, x1b);

  // y1 = relu(x1 @ w1^T + b1)  (bf16 out)
  gemm_wmma_kernel<<<dim3(D_FFN / 128, NTOK / 128), 256, 0, stream>>>(
      x1b, w1b, b1, nullptr, y1b, NTOK, D_FFN, D_MODEL, 1);

  // f2 = y1 @ w2^T + b2  (fp32 out)
  gemm_wmma_kernel<<<dim3(D_MODEL / 128, NTOK / 128), 256, 0, stream>>>(
      y1b, w2b, b2, f2, nullptr, NTOK, D_MODEL, D_FFN, 0);

  // out = LN2(x1 + f2)
  add_ln_kernel<<<NTOK, 256, 0, stream>>>(x1, f2, ln2a, ln2b, out, nullptr);
}
